// FilterTransU_8_46643344834870
// MI455X (gfx1250) — matrix operand order repaired and checked by reading, not compiled, not run on hardware
//
#include <hip/hip_runtime.h>
#include <hip/hip_bf16.h>
#include <math.h>
#include <stdint.h>

typedef __attribute__((ext_vector_type(16))) __bf16 bf16x16;
typedef __attribute__((ext_vector_type(8)))  __bf16 bf16x8;
typedef __attribute__((ext_vector_type(8)))  float  f32x8;

static __device__ __forceinline__ __bf16 tobf(float x) { return (__bf16)x; }

// fast sigmoid: v_exp_f32 + v_add + v_rcp_f32 (no IEEE div_scale fixup chain)
static __device__ __forceinline__ float fast_sigmoid(float x)
{
    return __builtin_amdgcn_rcpf(1.0f + __expf(-x));
}

// Build a 16-wide bf16 fragment from two contiguous 16B LDS chunks (ds_load_b128 x2).
static __device__ __forceinline__ bf16x16 ld_frag(const __bf16* p0, const __bf16* p1)
{
    bf16x8 lo = *(const bf16x8*)p0;
    bf16x8 hi = *(const bf16x8*)p1;
    return __builtin_shufflevector(lo, hi, 0, 1, 2, 3, 4, 5, 6, 7,
                                           8, 9, 10, 11, 12, 13, 14, 15);
}

// ---------------------------------------------------------------------------
// Generic fused WMMA GEMM:  out[m,n] = sum_k A(m,k) * W[n,k]  (+ epilogue)
// MODE (compile-time -> each instantiation carries only its own path):
//  0: A = x2 image gather [B,256,32,32]; epi: +bias[n] + pos2[(m%1024)*N+n]
//  1: A = x1 image gather, stride-2 pick; epi: +bias[n] + pos1[p*N+n]
//  2: A row-major [M,K]; epi: +bias[n] (if non-null) + extra[m*N+n] (if non-null)
//  3: A row-major [M,K]; epi: +bias[n]; store scattered to [B, N, 1024] (NCHW)
// Tile: 64(M) x 32(N) x 32(K), 4 waves, 2 WMMA per wave per k-step.
// LDS layouts give 2x ds_load_b128 per fragment:
//   As[row][k] row-contiguous (A frag: runs [kb..kb+7],[kb+16..kb+23])
//   Bs[n][k]   row-contiguous (B frag: runs [kb..kb+7],[kb+16..kb+23])
// ---------------------------------------------------------------------------
template <int MODE>
__global__ __launch_bounds__(128)
void ftu_wmma_gemm(const float* __restrict__ A, const float* __restrict__ W,
                   const float* __restrict__ bias, const float* __restrict__ extra,
                   float* __restrict__ out, int Nout, int Kdim)
{
    __shared__ __align__(16) __bf16 As[64][32];
    __shared__ __align__(16) __bf16 Bs[32][32];   // [n][k]

    const int tid  = threadIdx.x;
    const int lane = tid & 31;
    const int wv   = tid >> 5;
    const int n0   = blockIdx.x * 32;
    const int m0   = blockIdx.y * 64;

    f32x8 acc0 = {}; f32x8 acc1 = {};

    for (int k0 = 0; k0 < Kdim; k0 += 32) {
        // ---- stage A tile (64x32) ----
        if (MODE >= 2) {
            // row-major A: 2 float4 global loads + 1 ds_store_b128 per 8-elem unit
            for (int u = tid; u < 256; u += 128) {
                const int r = u >> 2, k8 = (u & 3) << 3;
                const float* src = A + (size_t)(m0 + r) * Kdim + (k0 + k8);
                const float4 f0 = ((const float4*)src)[0];
                const float4 f1 = ((const float4*)src)[1];
                bf16x8 v = { tobf(f0.x), tobf(f0.y), tobf(f0.z), tobf(f0.w),
                             tobf(f1.x), tobf(f1.y), tobf(f1.z), tobf(f1.w) };
                *(bf16x8*)&As[r][k8] = v;
            }
        } else {
            // channel-major image gather (small K), scalar staging
            for (int i = tid; i < 64 * 32; i += 128) {
                const int r = i >> 5, k = i & 31;
                const int gm = m0 + r, gk = k0 + k;
                const int b = gm >> 10, n = gm & 1023;
                float v;
                if (MODE == 0) {
                    v = A[((size_t)b * Kdim + gk) * 1024 + n];
                } else {
                    const int hq = n >> 5, wq = n & 31;
                    const int p = (hq * 2) * 64 + wq * 2;
                    v = A[((size_t)b * Kdim + gk) * 4096 + p];
                }
                As[r][k] = tobf(v);
            }
        }
        // ---- stage B tile: W[n][k] natural row-major, 1 unit per thread ----
        {
            const int u = tid;      // 128 units of 8 = 1024 elems
            const int n = u >> 2, k8 = (u & 3) << 3;
            const float* src = W + (size_t)(n0 + n) * Kdim + (k0 + k8);
            const float4 f0 = ((const float4*)src)[0];
            const float4 f1 = ((const float4*)src)[1];
            bf16x8 v = { tobf(f0.x), tobf(f0.y), tobf(f0.z), tobf(f0.w),
                         tobf(f1.x), tobf(f1.y), tobf(f1.z), tobf(f1.w) };
            *(bf16x8*)&Bs[n][k8] = v;
        }
        if (k0 + 32 < Kdim)
            __builtin_prefetch(&W[(size_t)n0 * Kdim + k0 + 32], 0, 1);
        __syncthreads();

        const int mrow = wv * 16 + (lane & 15);
        const int kb   = (lane < 16) ? 0 : 8;
        const int kb2  = (lane < 16) ? 0 : 16;
        const int nn   = lane & 15;

        bf16x16 aF = ld_frag(&As[mrow][kb], &As[mrow][kb + 16]);
        bf16x16 b0 = ld_frag(&Bs[nn][kb],      &Bs[nn][kb + 16]);
        bf16x16 b1 = ld_frag(&Bs[16 + nn][kb], &Bs[16 + nn][kb + 16]);

        acc0 = __builtin_amdgcn_wmma_f32_16x16x32_bf16(false, aF, false, b0, (short)0, acc0, false, false);
        acc1 = __builtin_amdgcn_wmma_f32_16x16x32_bf16(false, aF, false, b1, (short)0, acc1, false, false);
        __syncthreads();
    }

    // ---- store with fused epilogue (C/D: lane = col, VGPR r -> row r / r+8) ----
    const int nl = lane & 15;
    const int mb = (lane < 16) ? 0 : 8;
#pragma unroll
    for (int f = 0; f < 2; ++f) {
#pragma unroll
        for (int r = 0; r < 8; ++r) {
            const int gm = m0 + wv * 16 + mb + r;
            const int gn = n0 + f * 16 + nl;
            float v = (f == 0) ? acc0[r] : acc1[r];
            if (bias) v += bias[gn];
            if (MODE == 0) {
                const int n = gm & 1023;
                v += extra[(size_t)n * Nout + gn];            // pos2 [1024,N]
                out[(size_t)gm * Nout + gn] = v;
            } else if (MODE == 1) {
                const int n = gm & 1023;
                const int hq = n >> 5, wq = n & 31;
                const int p = (hq * 2) * 64 + wq * 2;
                v += extra[(size_t)p * Nout + gn];            // pos1 [4096,N]
                out[(size_t)gm * Nout + gn] = v;
            } else if (MODE == 2) {
                if (extra) v += extra[(size_t)gm * Nout + gn];
                out[(size_t)gm * Nout + gn] = v;
            } else {                                          // NCHW scatter
                const int b = gm >> 10, n = gm & 1023;
                out[((size_t)b * Nout + gn) * 1024 + n] = v;
            }
        }
    }
}

// ---------------------------------------------------------------------------
// LayerNorm over 384 channels, 1 block (128 thr) per row; optional relu.
// ---------------------------------------------------------------------------
__global__ __launch_bounds__(128)
void ftu_layernorm(const float* __restrict__ in, float* __restrict__ out,
                   const float* __restrict__ g, const float* __restrict__ bta, int relu)
{
    __shared__ float red[128];
    __shared__ float mu_s, rstd_s;
    const int row = blockIdx.x, tid = threadIdx.x;
    const float* r = in + (size_t)row * 384;
    const float x0 = r[tid], x1 = r[tid + 128], x2 = r[tid + 256];
    red[tid] = x0 + x1 + x2;
    __syncthreads();
    for (int s = 64; s > 0; s >>= 1) { if (tid < s) red[tid] += red[tid + s]; __syncthreads(); }
    if (tid == 0) mu_s = red[0] * (1.0f / 384.0f);
    __syncthreads();
    const float mu = mu_s;
    const float d0 = x0 - mu, d1 = x1 - mu, d2 = x2 - mu;
    red[tid] = d0 * d0 + d1 * d1 + d2 * d2;
    __syncthreads();
    for (int s = 64; s > 0; s >>= 1) { if (tid < s) red[tid] += red[tid + s]; __syncthreads(); }
    if (tid == 0) rstd_s = rsqrtf(red[0] * (1.0f / 384.0f) + 1e-5f);
    __syncthreads();
    const float rs = rstd_s;
    float* o = out + (size_t)row * 384;
    float y0 = d0 * rs * g[tid]       + bta[tid];
    float y1 = d1 * rs * g[tid + 128] + bta[tid + 128];
    float y2 = d2 * rs * g[tid + 256] + bta[tid + 256];
    if (relu) { y0 = fmaxf(y0, 0.f); y1 = fmaxf(y1, 0.f); y2 = fmaxf(y2, 0.f); }
    o[tid] = y0; o[tid + 128] = y1; o[tid + 256] = y2;
}

// ---------------------------------------------------------------------------
// Spectral filter: per (b,c), y = irfft2( rfft2(x32x32) * wc ), norm='ortho'.
// Input grid is pulled into LDS with GLOBAL_LOAD_ASYNC_TO_LDS_B32 (CDNA5 async
// copy: no VGPR round-trip, tracked by ASYNCcnt). 4 table-driven DFT passes;
// intermediates stored transposed so every reduction axis is contiguous
// (float4 LDS reads); float2 twiddle table -> one ds_load_b64 per MAC pair.
// ---------------------------------------------------------------------------
__global__ __launch_bounds__(256)
void ftu_spectral(const float* __restrict__ src, float* __restrict__ dst,
                  const float* __restrict__ cw)
{
    __shared__ __align__(16) float Ar[1024], Ai[1024], Br[1024], Bi[1024];
    __shared__ float2 tw[32];
    const int b = blockIdx.x / 384;
    const int c = blockIdx.x % 384;
    const int tid = threadIdx.x;

    if (tid < 32) {
        const float th = -6.2831853071795864769f * (float)tid / 32.0f;
        tw[tid] = make_float2(cosf(th), sinf(th));
    }
    // async gather of x[h][w] (stride-384 channel gather) straight into LDS
    for (int i = tid; i < 1024; i += 256) {
        const unsigned long long gaddr =
            (unsigned long long)(uintptr_t)(src + ((size_t)(b * 1024 + i)) * 384 + c);
        const unsigned int laddr = (unsigned int)(uintptr_t)(&Ar[i]);
        asm volatile("global_load_async_to_lds_b32 %0, %1, off"
                     :: "v"(laddr), "v"(gaddr) : "memory");
    }
    asm volatile("s_wait_asynccnt 0" ::: "memory");
    __syncthreads();

    // pass 1: DFT along w (real input). out T stored transposed: Br/Bi[u][h]
    for (int id = tid; id < 1024; id += 256) {
        const int h = id >> 5, u = id & 31;
        float sr = 0.f, si = 0.f;
        int kidx = 0;
        for (int w4 = 0; w4 < 32; w4 += 4) {
            const float4 x = *(const float4*)&Ar[h * 32 + w4];
            float2 t;
            t = tw[kidx]; sr += x.x * t.x; si += x.x * t.y; kidx = (kidx + u) & 31;
            t = tw[kidx]; sr += x.y * t.x; si += x.y * t.y; kidx = (kidx + u) & 31;
            t = tw[kidx]; sr += x.z * t.x; si += x.z * t.y; kidx = (kidx + u) & 31;
            t = tw[kidx]; sr += x.w * t.x; si += x.w * t.y; kidx = (kidx + u) & 31;
        }
        Br[u * 32 + h] = sr; Bi[u * 32 + h] = si;
    }
    __syncthreads();
    // pass 2: DFT along h. in: Br/Bi[uw][h]; out Z transposed: Ar/Ai[uw][uh]
    for (int id = tid; id < 1024; id += 256) {
        const int uw = id >> 5, uh = id & 31;
        float sr = 0.f, si = 0.f;
        int kidx = 0;
        for (int h4 = 0; h4 < 32; h4 += 4) {
            const float4 xr = *(const float4*)&Br[uw * 32 + h4];
            const float4 xi = *(const float4*)&Bi[uw * 32 + h4];
            float2 t;
            t = tw[kidx]; sr += xr.x * t.x - xi.x * t.y; si += xr.x * t.y + xi.x * t.x; kidx = (kidx + uh) & 31;
            t = tw[kidx]; sr += xr.y * t.x - xi.y * t.y; si += xr.y * t.y + xi.y * t.x; kidx = (kidx + uh) & 31;
            t = tw[kidx]; sr += xr.z * t.x - xi.z * t.y; si += xr.z * t.y + xi.z * t.x; kidx = (kidx + uh) & 31;
            t = tw[kidx]; sr += xr.w * t.x - xi.w * t.y; si += xr.w * t.y + xi.w * t.x; kidx = (kidx + uh) & 31;
        }
        Ar[uw * 32 + uh] = sr; Ai[uw * 32 + uh] = si;
    }
    __syncthreads();
    // pointwise Hermitian-extended weight, in [uw][uh] layout
    for (int id = tid; id < 1024; id += 256) {
        const int uw = id >> 5, uh = id & 31;
        float wr, wi;
        if (uw <= 16) {
            const size_t ix = ((size_t)(uh * 17 + uw) * 384 + c) * 2;
            wr = cw[ix]; wi = cw[ix + 1];
        } else {
            const int u2 = (32 - uh) & 31, v2 = 32 - uw;
            const size_t ix = ((size_t)(u2 * 17 + v2) * 384 + c) * 2;
            wr = cw[ix]; wi = -cw[ix + 1];
        }
        const float xr = Ar[id], xi = Ai[id];
        Br[id] = xr * wr - xi * wi; Bi[id] = xr * wi + xi * wr;
    }
    __syncthreads();
    // pass 3: inverse DFT along h (conj). in: Br/Bi[uw][u]; out G: Ar/Ai[h][uw]
    for (int id = tid; id < 1024; id += 256) {
        const int h = id >> 5, uw = id & 31;
        float sr = 0.f, si = 0.f;
        int kidx = 0;
        for (int u4 = 0; u4 < 32; u4 += 4) {
            const float4 xr = *(const float4*)&Br[uw * 32 + u4];
            const float4 xi = *(const float4*)&Bi[uw * 32 + u4];
            float2 t;
            t = tw[kidx]; sr += xr.x * t.x + xi.x * t.y; si += xi.x * t.x - xr.x * t.y; kidx = (kidx + h) & 31;
            t = tw[kidx]; sr += xr.y * t.x + xi.y * t.y; si += xi.y * t.x - xr.y * t.y; kidx = (kidx + h) & 31;
            t = tw[kidx]; sr += xr.z * t.x + xi.z * t.y; si += xi.z * t.x - xr.z * t.y; kidx = (kidx + h) & 31;
            t = tw[kidx]; sr += xr.w * t.x + xi.w * t.y; si += xi.w * t.x - xr.w * t.y; kidx = (kidx + h) & 31;
        }
        Ar[h * 32 + uw] = sr; Ai[h * 32 + uw] = si;
    }
    __syncthreads();
    // pass 4: inverse DFT along w, real part, 1/1024 ortho scale
    for (int id = tid; id < 1024; id += 256) {
        const int h = id >> 5, w = id & 31;
        float acc = 0.f;
        int kidx = 0;
        for (int v4 = 0; v4 < 32; v4 += 4) {
            const float4 gr = *(const float4*)&Ar[h * 32 + v4];
            const float4 gi = *(const float4*)&Ai[h * 32 + v4];
            float2 t;
            t = tw[kidx]; acc += gr.x * t.x + gi.x * t.y; kidx = (kidx + w) & 31;
            t = tw[kidx]; acc += gr.y * t.x + gi.y * t.y; kidx = (kidx + w) & 31;
            t = tw[kidx]; acc += gr.z * t.x + gi.z * t.y; kidx = (kidx + w) & 31;
            t = tw[kidx]; acc += gr.w * t.x + gi.w * t.y; kidx = (kidx + w) & 31;
        }
        dst[((size_t)(b * 1024 + h * 32 + w)) * 384 + c] = acc * (1.0f / 1024.0f);
    }
}

// ---------------------------------------------------------------------------
// Streaming sigmoid attention, K==V. One block = (b, head, 64 q-rows), 4 waves.
// Kt[t][d]  : row-contiguous -> score B-fragments are 2x ds_load_b128
// KtT[d][t] : row-contiguous -> ctx  B-fragments are 2x ds_load_b128
// Swb (bf16): per-wave C->A transpose scratch, P fragment 2x ds_load_b128
// Distinct fragment variables let the scheduler issue ds_load_b128 batches
// ahead of the WMMAs with partial s_wait_dscnt overlap.
// ---------------------------------------------------------------------------
__global__ __launch_bounds__(128)
void ftu_attention(const float* __restrict__ qg, const float* __restrict__ xg,
                   float* __restrict__ ctx)
{
    __shared__ __align__(16) __bf16 Kt[32][96];
    __shared__ __align__(16) __bf16 KtT[96][32];
    __shared__ __align__(16) __bf16 Swb[4][16][32];

    const int tid  = threadIdx.x;
    const int lane = tid & 31;
    const int wv   = tid >> 5;
    const int qbase = blockIdx.x * 64;
    const int head  = blockIdx.y;
    const int b     = blockIdx.z;
    const float scale = rsqrtf(96.0f);

    // Q fragments (loop-invariant): 3 chunks of K=32 over dh=96
    bf16x16 aQ[3];
    {
        const int qrow = qbase + wv * 16 + (lane & 15);
        const int kb = (lane < 16) ? 0 : 8;
        const float* qp = qg + ((size_t)(b * 1024 + qrow)) * 384 + head;
#pragma unroll
        for (int kc = 0; kc < 3; ++kc) {
#pragma unroll
            for (int e = 0; e < 16; ++e) {
                const int d = kc * 32 + kb + e + ((e >= 8) ? 8 : 0);
                aQ[kc][e] = tobf(qp[d * 4] * scale);
            }
        }
    }

    f32x8 acc[6];
#pragma unroll
    for (int i = 0; i < 6; ++i) acc[i] = (f32x8){};

    const int nl = lane & 15;
    const int mb = (lane < 16) ? 0 : 8;
    const int kb  = (lane < 16) ? 0 : 8;
    const int kb2 = (lane < 16) ? 0 : 16;

    for (int j0 = 0; j0 < 1024; j0 += 32) {
        __syncthreads();
        // stage K/V tile (and transpose) as bf16, 16B LDS stores
        for (int u = tid; u < 384; u += 128) {             // Kt: (t, d8)
            const int t = u / 12, d8 = (u % 12) * 8;
            const float* p = xg + ((size_t)(b * 1024 + j0 + t)) * 384 + head;
            bf16x8 v;
#pragma unroll
            for (int i = 0; i < 8; ++i) v[i] = tobf(p[(d8 + i) * 4]);
            *(bf16x8*)&Kt[t][d8] = v;
        }
        for (int u = tid; u < 384; u += 128) {             // KtT: (d, t8)
            const int d = u >> 2, t8 = (u & 3) << 3;
            const float* p = xg + ((size_t)(b * 1024 + j0 + t8)) * 384 + d * 4 + head;
            bf16x8 v;
#pragma unroll
            for (int i = 0; i < 8; ++i) v[i] = tobf(p[(size_t)i * 384]);
            *(bf16x8*)&KtT[d][t8] = v;
        }
        __syncthreads();

        // scores S[16q, 32kv] in two 16-col halves, accumulated over dh
#pragma unroll
        for (int sf = 0; sf < 2; ++sf) {
            const int tt = sf * 16 + (lane & 15);
            // load all three B fragments first (distinct regs -> overlapped ds)
            bf16x16 kB0 = ld_frag(&Kt[tt][kb],      &Kt[tt][kb + 16]);
            bf16x16 kB1 = ld_frag(&Kt[tt][32 + kb], &Kt[tt][32 + kb + 16]);
            bf16x16 kB2 = ld_frag(&Kt[tt][64 + kb], &Kt[tt][64 + kb + 16]);
            f32x8 s = {};
            s = __builtin_amdgcn_wmma_f32_16x16x32_bf16(false, aQ[0], false, kB0, (short)0, s, false, false);
            s = __builtin_amdgcn_wmma_f32_16x16x32_bf16(false, aQ[1], false, kB1, (short)0, s, false, false);
            s = __builtin_amdgcn_wmma_f32_16x16x32_bf16(false, aQ[2], false, kB2, (short)0, s, false, false);
#pragma unroll
            for (int r = 0; r < 8; ++r)
                Swb[wv][mb + r][sf * 16 + nl] = tobf(fast_sigmoid(s[r]));
        }
        __syncthreads();

        // P fragment (A layout 16x32) from wave-private scratch
        bf16x16 pF = ld_frag(&Swb[wv][lane & 15][kb], &Swb[wv][lane & 15][kb + 16]);

        // ctx[16q, 96d] += P[16,32] * V[32,96]  (V == K tile, via KtT);
        // process output columns in pairs so both fragments' loads overlap
#pragma unroll
        for (int nf = 0; nf < 6; nf += 2) {
            const int dc0 = nf * 16 + (lane & 15);
            const int dc1 = (nf + 1) * 16 + (lane & 15);
            bf16x16 vB0 = ld_frag(&KtT[dc0][kb], &KtT[dc0][kb + 16]);
            bf16x16 vB1 = ld_frag(&KtT[dc1][kb], &KtT[dc1][kb + 16]);
            acc[nf]     = __builtin_amdgcn_wmma_f32_16x16x32_bf16(false, pF, false, vB0, (short)0, acc[nf],     false, false);
            acc[nf + 1] = __builtin_amdgcn_wmma_f32_16x16x32_bf16(false, pF, false, vB1, (short)0, acc[nf + 1], false, false);
        }
    }

    // write ctx back in '(dim_head heads)' channel packing
#pragma unroll
    for (int nf = 0; nf < 6; ++nf) {
#pragma unroll
        for (int r = 0; r < 8; ++r) {
            const int d = nf * 16 + nl;
            const int gq = b * 1024 + qbase + wv * 16 + mb + r;
            ctx[(size_t)gq * 384 + d * 4 + head] = acc[nf][r];
        }
    }
}

// ---------------------------------------------------------------------------
extern "C" void kernel_launch(void* const* d_in, const int* in_sizes, int n_in,
                              void* d_out, int out_size, void* d_ws, size_t ws_size,
                              hipStream_t stream)
{
    const float* x1     = (const float*)d_in[0];
    const float* x2     = (const float*)d_in[1];
    const float* emb1_w = (const float*)d_in[2];
    const float* emb1_b = (const float*)d_in[3];
    const float* pos1   = (const float*)d_in[4];
    const float* emb2_w = (const float*)d_in[5];
    const float* emb2_b = (const float*)d_in[6];
    const float* pos2   = (const float*)d_in[7];
    const float* ln_g   = (const float*)d_in[8];
    const float* ln_b   = (const float*)d_in[9];
    const float* cw     = (const float*)d_in[10];
    const float* out_w  = (const float*)d_in[11];
    const float* out_b  = (const float*)d_in[12];
    const float* mlp_w  = (const float*)d_in[13];
    const float* conv_w = (const float*)d_in[14];
    const float* conv_b = (const float*)d_in[15];
    float* out = (float*)d_out;

    const size_t S = (size_t)8192 * 384;
    float* b0 = (float*)d_ws;      // x2e residue
    float* b1 = b0 + S;            // x_in -> out1 (residue2)
    float* b2 = b1 + S;            // xg   -> out2
    float* b3 = b2 + S;            // q tokens -> q_in -> relu(ln(out1))
    float* b4 = b3 + S;            // qg
    float* b5 = b4 + S;            // ctx

    const dim3 blk128(128), blk256(256);

    // 1. x2e = proj(x2) + emb2_b + pos2  -> b0
    ftu_wmma_gemm<0><<<dim3(12, 128), blk128, 0, stream>>>(x2, emb2_w, emb2_b, pos2, b0, 384, 256);
    // 2. q tokens (downsample-fused proj(x1) + emb1_b + pos1) -> b3
    ftu_wmma_gemm<1><<<dim3(12, 128), blk128, 0, stream>>>(x1, emb1_w, emb1_b, pos1, b3, 384, 128);
    // 3. x_in = LN(x2e) -> b1 ; q_in = LN(q) -> b3 (in place)
    ftu_layernorm<<<8192, blk128, 0, stream>>>(b0, b1, ln_g, ln_b, 0);
    ftu_layernorm<<<8192, blk128, 0, stream>>>(b3, b3, ln_g, ln_b, 0);
    // 4. spectral filters: xg -> b2, qg -> b4
    ftu_spectral<<<3072, blk256, 0, stream>>>(b1, b2, cw);
    ftu_spectral<<<3072, blk256, 0, stream>>>(b3, b4, cw);
    // 5. sigmoid attention -> ctx b5
    ftu_attention<<<dim3(16, 4, 8), blk128, 0, stream>>>(b4, b2, b5);
    // 6. out1 = ctx @ out_w^T + out_b + residue(b0) -> b1
    ftu_wmma_gemm<2><<<dim3(12, 128), blk128, 0, stream>>>(b5, out_w, out_b, b0, b1, 384, 384);
    // 7. relu(LN(out1)) -> b3
    ftu_layernorm<<<8192, blk128, 0, stream>>>(b1, b3, ln_g, ln_b, 1);
    // 8. out2 = b3 @ mlp_w^T + out1(b1) -> b2
    ftu_wmma_gemm<2><<<dim3(12, 128), blk128, 0, stream>>>(b3, mlp_w, nullptr, b1, b2, 384, 384);
    // 9. final 1x1 conv, NCHW scatter -> d_out
    ftu_wmma_gemm<3><<<dim3(8, 128), blk128, 0, stream>>>(b2, conv_w, conv_b, nullptr, out, 256, 384);
}
